// DynamicSparseRouter_42872363549114
// MI455X (gfx1250) — compile-verified, matrix operand order checked by reading, not run on hardware
//
#include <hip/hip_runtime.h>
#include <hip/hip_bf16.h>
#include <math.h>

typedef __bf16 bf16_t;
typedef __attribute__((ext_vector_type(16))) __bf16 v16bf;
typedef __attribute__((ext_vector_type(8)))  float  v8f;

#define BN_TOK 16384   // B*N tokens
#define SEQ_N  4096
#define DIM    1024
#define HEADS  8
#define DHEAD  128
#define KSLOT  64
#define WIN    32
#define RANKLR 32

// ---------- helpers ----------
__device__ __forceinline__ bf16_t f2bf(float f) {
  unsigned u = __float_as_uint(f);
  u += 0x7fffu + ((u >> 16) & 1u);              // round-to-nearest-even
  unsigned short hs = (unsigned short)(u >> 16);
  bf16_t r;
  __builtin_memcpy(&r, &hs, sizeof(r));
  return r;
}

// Load one 16x32 bf16 WMMA operand fragment (A layout; B uses the same layout
// on a transposed (N x Kc) matrix). Per CDNA5 ISA 7.12.2:
//   lane m = lane%16 selects the row; half h = lane/16;
//   elements 0..7  = K in [k0+8h,   k0+8h+8)   (contiguous, 16 bytes)
//   elements 8..15 = K in [k0+16+8h,k0+16+8h+8)
__device__ __forceinline__ v16bf load_frag(const bf16_t* __restrict__ tile,
                                           int ld, int k0, int lane) {
  const int m = lane & 15;
  const int h = lane >> 4;
  const bf16_t* p = tile + (size_t)m * ld + k0 + 8 * h;
  union { uint4 u[2]; v16bf v; } c;
  c.u[0] = *(const uint4*)(p);
  c.u[1] = *(const uint4*)(p + 16);
  return c.v;
}

// ---------- generic bf16 WMMA GEMM: C = epilogue(alpha*A*Bt^T + bias) ----------
// A: (M x Kc) row-major bf16, Bt: (N x Kc) row-major bf16 (i.e. B transposed).
// grid.z batches via element-offset strides (aBS/bBS/cBS).
// EP: 0=none, 1=clamped sigmoid, 2=exact gelu.
template <int WM, int WN, int WAVES_M, int WAVES_N, int EP, bool OUTBF>
__global__ void __launch_bounds__(WAVES_M * WAVES_N * 32)
gemm_bf16(const bf16_t* __restrict__ A, const bf16_t* __restrict__ Bt,
          const float* __restrict__ bias, void* __restrict__ Cv,
          int Kc, int lda, int ldb, int ldc, float alpha,
          long long aBS, long long bBS, long long cBS) {
  const int lane = threadIdx.x & 31;
  const int wave = threadIdx.x >> 5;
  const int wm = wave % WAVES_M;
  const int wn = wave / WAVES_M;

  A  += (size_t)blockIdx.z * (size_t)aBS;
  Bt += (size_t)blockIdx.z * (size_t)bBS;
  float*  Cf = (float*)Cv  + (size_t)blockIdx.z * (size_t)cBS;
  bf16_t* Cb = (bf16_t*)Cv + (size_t)blockIdx.z * (size_t)cBS;

  const int tileM0 = blockIdx.x * (WAVES_M * WM * 16) + wm * WM * 16;
  const int tileN0 = blockIdx.y * (WAVES_N * WN * 16) + wn * WN * 16;

  const v8f vzero = {0.f, 0.f, 0.f, 0.f, 0.f, 0.f, 0.f, 0.f};
  v8f acc[WM][WN];
#pragma unroll
  for (int i = 0; i < WM; ++i)
#pragma unroll
    for (int j = 0; j < WN; ++j) acc[i][j] = vzero;

  for (int k0 = 0; k0 < Kc; k0 += 32) {
    v16bf afrag[WM], bfrag[WN];
#pragma unroll
    for (int i = 0; i < WM; ++i)
      afrag[i] = load_frag(A + (size_t)(tileM0 + i * 16) * lda, lda, k0, lane);
#pragma unroll
    for (int j = 0; j < WN; ++j)
      bfrag[j] = load_frag(Bt + (size_t)(tileN0 + j * 16) * ldb, ldb, k0, lane);
#pragma unroll
    for (int i = 0; i < WM; ++i)
#pragma unroll
      for (int j = 0; j < WN; ++j)
        acc[i][j] = __builtin_amdgcn_wmma_f32_16x16x32_bf16(
            false, afrag[i], false, bfrag[j], (short)0, acc[i][j], false, false);
  }

  // C/D layout: VGPR r -> row r + 8*(lane/16), col = lane%16
  const int h = lane >> 4;
  const int nl = lane & 15;
#pragma unroll
  for (int j = 0; j < WN; ++j) {
    const int col = tileN0 + j * 16 + nl;
    const float bv = bias ? bias[col] : 0.f;
#pragma unroll
    for (int i = 0; i < WM; ++i) {
#pragma unroll
      for (int r = 0; r < 8; ++r) {
        const int rowg = tileM0 + i * 16 + h * 8 + r;
        float v = acc[i][j][r] * alpha + bv;
        if (EP == 1) v = fmaxf(1.f / (1.f + expf(-v)), 1e-4f);
        if (EP == 2) v = 0.5f * v * (1.f + erff(v * 0.70710678118654752f));
        const size_t idx = (size_t)rowg * ldc + col;
        if constexpr (OUTBF) Cb[idx] = f2bf(v); else Cf[idx] = v;
      }
    }
  }
}

// ---------- small prep kernels ----------
__global__ void k_transpose_bf16(const float* __restrict__ src,
                                 bf16_t* __restrict__ dst, int R, int C) {
  int idx = blockIdx.x * blockDim.x + threadIdx.x;
  if (idx >= R * C) return;
  int r = idx / C, c = idx % C;
  dst[(size_t)c * R + r] = f2bf(src[idx]);
}

__global__ void k_convert_bf16(const float* __restrict__ src,
                               bf16_t* __restrict__ dst, int n) {
  int idx = blockIdx.x * blockDim.x + threadIdx.x;
  if (idx < n) dst[idx] = f2bf(src[idx]);
}

// V = slot_keys @ Wsv + bsv, stored transposed (DIM x KSLOT) bf16 so the
// routed GEMM's B operand loads with the standard fragment pattern.
__global__ void k_slot_values(const float* __restrict__ sk,
                              const float* __restrict__ Wsv,
                              const float* __restrict__ bsv,
                              bf16_t* __restrict__ vt) {
  int idx = blockIdx.x * blockDim.x + threadIdx.x;   // KSLOT*DIM threads
  int k = idx >> 10;
  int d = idx & (DIM - 1);
  float acc = bsv[d];
  const float* srow = sk + (size_t)k * DIM;
  for (int i = 0; i < DIM; ++i) acc += srow[i] * Wsv[(size_t)i * DIM + d];
  vt[(size_t)d * KSLOT + k] = f2bf(acc);
}

// causal windowed mean of x, plus bf16 copies of x and xw
__global__ void k_windowed_mean(const float* __restrict__ x,
                                bf16_t* __restrict__ x_bf,
                                bf16_t* __restrict__ xw_bf) {
  const int bn = blockIdx.x;                 // token index in [0, B*N)
  const int t = bn & (SEQ_N - 1);
  const int d0 = threadIdx.x * 4;
  const float* row = x + (size_t)bn * DIM;
  const float4 v = *(const float4*)(row + d0);
  bf16_t* xb = x_bf + (size_t)bn * DIM + d0;
  xb[0] = f2bf(v.x); xb[1] = f2bf(v.y); xb[2] = f2bf(v.z); xb[3] = f2bf(v.w);
  const int cnt = (t + 1 < WIN) ? (t + 1) : WIN;
  float sx = 0.f, sy = 0.f, sz = 0.f, sw = 0.f;
  for (int w = 0; w < cnt; ++w) {
    const float4 u = *(const float4*)(row - (size_t)w * DIM + d0);
    sx += u.x; sy += u.y; sz += u.z; sw += u.w;
  }
  const float inv = 1.f / (float)cnt;
  bf16_t* xwb = xw_bf + (size_t)bn * DIM + d0;
  xwb[0] = f2bf(sx * inv); xwb[1] = f2bf(sy * inv);
  xwb[2] = f2bf(sz * inv); xwb[3] = f2bf(sw * inv);
}

// s = gate*(scaled_score + offset) + pos_bias; softmax over K=64; bf16 weights.
// One wave32 per (head, token) row; 2 slots per lane; shuffle reductions.
__global__ void k_combine_softmax(const float* __restrict__ scores,  // (H,BN,K)
                                  const float* __restrict__ gate,    // (BN,K)
                                  const float* __restrict__ offs,    // (BN,K)
                                  const float* __restrict__ pos_bias,// (N,K)
                                  bf16_t* __restrict__ wout) {       // (H,BN,K)
  const int lane = threadIdx.x & 31;
  const int wave = threadIdx.x >> 5;
  const int row = blockIdx.x * 8 + wave;     // 0 .. H*BN-1
  const int bn = row % BN_TOK;
  const int t = bn & (SEQ_N - 1);
  const float* sc = scores + (size_t)row * KSLOT;
  const float* g  = gate   + (size_t)bn * KSLOT;
  const float* o  = offs   + (size_t)bn * KSLOT;
  const float* pb = pos_bias + (size_t)t * KSLOT;
  const int k0 = lane, k1 = lane + 32;
  float s0 = g[k0] * (sc[k0] + o[k0]) + pb[k0];
  float s1 = g[k1] * (sc[k1] + o[k1]) + pb[k1];
  float m = fmaxf(s0, s1);
  for (int d = 16; d > 0; d >>= 1) m = fmaxf(m, __shfl_xor(m, d, 32));
  float e0 = expf(s0 - m), e1 = expf(s1 - m);
  float s = e0 + e1;
  for (int d = 16; d > 0; d >>= 1) s += __shfl_xor(s, d, 32);
  const float inv = 1.f / s;
  bf16_t* wr = wout + (size_t)row * KSLOT;
  wr[k0] = f2bf(e0 * inv);
  wr[k1] = f2bf(e1 * inv);
}

// ---------- orchestration ----------
extern "C" void kernel_launch(void* const* d_in, const int* in_sizes, int n_in,
                              void* d_out, int out_size, void* d_ws, size_t ws_size,
                              hipStream_t stream) {
  const float* x         = (const float*)d_in[0];
  const float* Wq        = (const float*)d_in[1];
  const float* bq        = (const float*)d_in[2];
  const float* slot_keys = (const float*)d_in[3];
  const float* Wsv       = (const float*)d_in[4];
  const float* bsv       = (const float*)d_in[5];
  const float* Wg        = (const float*)d_in[6];
  const float* bg        = (const float*)d_in[7];
  const float* pos_bias  = (const float*)d_in[8];
  const float* Wdown     = (const float*)d_in[9];
  const float* bdown     = (const float*)d_in[10];
  const float* Wup       = (const float*)d_in[11];
  const float* bup       = (const float*)d_in[12];
  const float* Wo        = (const float*)d_in[13];
  const float* bo        = (const float*)d_in[14];
  float* out = (float*)d_out;

  char* ws = (char*)d_ws;
  size_t off = 0;
  auto alloc = [&](size_t bytes) -> void* {
    void* p = ws + off;
    off = (off + bytes + 255) & ~(size_t)255;
    return p;
  };

  bf16_t* wqT    = (bf16_t*)alloc((size_t)DIM * DIM * 2);       // (Dout,Din)
  bf16_t* woT    = (bf16_t*)alloc((size_t)DIM * DIM * 2);
  bf16_t* wgT    = (bf16_t*)alloc((size_t)KSLOT * DIM * 2);
  bf16_t* wdownT = (bf16_t*)alloc((size_t)RANKLR * DIM * 2);
  bf16_t* wupT   = (bf16_t*)alloc((size_t)KSLOT * RANKLR * 2);
  bf16_t* sk_bf  = (bf16_t*)alloc((size_t)KSLOT * DIM * 2);
  bf16_t* vt_bf  = (bf16_t*)alloc((size_t)DIM * KSLOT * 2);     // V^T (D,K)
  bf16_t* x_bf   = (bf16_t*)alloc((size_t)BN_TOK * DIM * 2);
  bf16_t* xw_bf  = (bf16_t*)alloc((size_t)BN_TOK * DIM * 2);
  bf16_t* q_bf   = (bf16_t*)alloc((size_t)BN_TOK * DIM * 2);
  bf16_t* down_bf= (bf16_t*)alloc((size_t)BN_TOK * RANKLR * 2);
  bf16_t* rout_bf= (bf16_t*)alloc((size_t)BN_TOK * DIM * 2);
  bf16_t* w_bf   = (bf16_t*)alloc((size_t)HEADS * BN_TOK * KSLOT * 2);
  float*  gate_f = (float*)alloc((size_t)BN_TOK * KSLOT * 4);
  float*  offs_f = (float*)alloc((size_t)BN_TOK * KSLOT * 4);
  float*  score_f= (float*)alloc((size_t)HEADS * BN_TOK * KSLOT * 4);

  const float SCALE = 0.08838834764831845f;   // 1/sqrt(128)

  // --- prep ---
  k_transpose_bf16<<<(DIM * DIM + 255) / 256, 256, 0, stream>>>(Wq, wqT, DIM, DIM);
  k_transpose_bf16<<<(DIM * DIM + 255) / 256, 256, 0, stream>>>(Wo, woT, DIM, DIM);
  k_transpose_bf16<<<(DIM * KSLOT + 255) / 256, 256, 0, stream>>>(Wg, wgT, DIM, KSLOT);
  k_transpose_bf16<<<(DIM * RANKLR + 255) / 256, 256, 0, stream>>>(Wdown, wdownT, DIM, RANKLR);
  k_transpose_bf16<<<(RANKLR * KSLOT + 255) / 256, 256, 0, stream>>>(Wup, wupT, RANKLR, KSLOT);
  k_convert_bf16<<<(KSLOT * DIM + 255) / 256, 256, 0, stream>>>(slot_keys, sk_bf, KSLOT * DIM);
  k_slot_values<<<(KSLOT * DIM + 255) / 256, 256, 0, stream>>>(slot_keys, Wsv, bsv, vt_bf);
  k_windowed_mean<<<BN_TOK, 256, 0, stream>>>(x, x_bf, xw_bf);

  // --- G1: Q = x @ Wq + bq  (bf16 out) ---
  gemm_bf16<4, 2, 2, 4, 0, true><<<dim3(BN_TOK / 128, DIM / 128, 1), 256, 0, stream>>>(
      x_bf, wqT, bq, q_bf, DIM, DIM, DIM, DIM, 1.f, 0, 0, 0);

  // --- G2: gate = clamp(sigmoid(xw @ Wg + bg)) (f32 out) ---
  gemm_bf16<4, 2, 4, 2, 1, false><<<dim3(BN_TOK / 256, 1, 1), 256, 0, stream>>>(
      xw_bf, wgT, bg, gate_f, DIM, DIM, DIM, KSLOT, 1.f, 0, 0, 0);

  // --- G3: down = gelu(xw @ Wdown + bdown) (bf16 out) ---
  gemm_bf16<2, 2, 8, 1, 2, true><<<dim3(BN_TOK / 256, 1, 1), 256, 0, stream>>>(
      xw_bf, wdownT, bdown, down_bf, DIM, DIM, DIM, RANKLR, 1.f, 0, 0, 0);

  // --- G4: offset = down @ Wup + bup (f32 out) ---
  gemm_bf16<4, 2, 4, 2, 0, false><<<dim3(BN_TOK / 256, 1, 1), 256, 0, stream>>>(
      down_bf, wupT, bup, offs_f, RANKLR, RANKLR, RANKLR, KSLOT, 1.f, 0, 0, 0);

  // --- G5: per-head scores = SCALE * Q_h @ slot_keys_h^T (f32 out, batched) ---
  gemm_bf16<4, 2, 4, 2, 0, false><<<dim3(BN_TOK / 256, 1, HEADS), 256, 0, stream>>>(
      q_bf, sk_bf, nullptr, score_f, DHEAD, DIM, DIM, KSLOT, SCALE,
      (long long)DHEAD, (long long)DHEAD, (long long)BN_TOK * KSLOT);

  // --- combine + softmax -> routing weights (bf16) ---
  k_combine_softmax<<<(HEADS * BN_TOK) / 8, 256, 0, stream>>>(
      score_f, gate_f, offs_f, pos_bias, w_bf);

  // --- G6: routed_h = w_h @ V_h (bf16 out, batched over heads) ---
  gemm_bf16<4, 2, 2, 4, 0, true><<<dim3(BN_TOK / 128, 1, HEADS), 256, 0, stream>>>(
      w_bf, vt_bf, nullptr, rout_bf, KSLOT, KSLOT, KSLOT, DIM, 1.f,
      (long long)BN_TOK * KSLOT, (long long)DHEAD * KSLOT, (long long)DHEAD);

  // --- G7: out = routed @ Wo + bo (f32 out) ---
  gemm_bf16<4, 2, 2, 4, 0, false><<<dim3(BN_TOK / 128, DIM / 128, 1), 256, 0, stream>>>(
      rout_bf, woT, bo, out, DIM, DIM, DIM, DIM, 1.f, 0, 0, 0);
}
